// MambaDecoderLayer_60541859004421
// MI455X (gfx1250) — matrix operand order repaired and checked by reading, not compiled, not run on hardware
//
#include <hip/hip_runtime.h>

typedef __bf16 bf16;
typedef __bf16 v16bf __attribute__((ext_vector_type(16)));
typedef __bf16 v8bf  __attribute__((ext_vector_type(8)));
typedef float  v8f   __attribute__((ext_vector_type(8)));

#define D_MODEL 1024
#define D_INNER 2048
#define D_STATE 16
#define DT_RANK 64
#define XDBL_N  96
#define NB      4
#define LSEQ    1024
#define NROWS   (NB * LSEQ)   /* 4096 */

#if __has_builtin(__builtin_amdgcn_tensor_load_to_lds) && __has_builtin(__builtin_amdgcn_s_wait_tensorcnt)
#define USE_TDM 1
#else
#define USE_TDM 0
#endif

// ---------------------------------------------------------------------------
// small helpers
// ---------------------------------------------------------------------------
__device__ inline v8f v8f_zero() {
  v8f z;
#pragma unroll
  for (int i = 0; i < 8; ++i) z[i] = 0.0f;
  return z;
}
__device__ inline float silu_f(float x) { return x / (1.0f + expf(-x)); }
__device__ inline float softplus_f(float x) { return (x > 20.0f) ? x : log1pf(expf(x)); }
__device__ inline float gelu_f(float x) { return 0.5f * x * (1.0f + erff(x * 0.70710678118654752f)); }

// A-matrix 16x32 bf16 fragment (ISA 7.12.2): lanes 0-15: M=lane, K=0..7 & 16..23;
// lanes 16-31: M=lane-16, K=8..15 & 24..31.
__device__ inline v16bf load_fragA(const bf16* tile, int rowBase, int stride, int kOff) {
  int lane = threadIdx.x & 31;
  const bf16* p = tile + (rowBase + (lane & 15)) * stride + kOff + ((lane & 16) ? 8 : 0);
  v8bf lo = *(const v8bf*)p;
  v8bf hi = *(const v8bf*)(p + 16);
  v16bf f;
#pragma unroll
  for (int i = 0; i < 8; ++i) { f[i] = lo[i]; f[i + 8] = hi[i]; }
  return f;
}
// B-matrix 32x16 bf16 fragment: lane&15 = N column; lanes 0-15 hold K=0..7 & 16..23,
// lanes 16-31 hold K=8..15 & 24..31 (same K order as the A fragment).
__device__ inline v16bf load_fragB(const bf16* tile, int rowBase, int stride, int kOff) {
  int lane = threadIdx.x & 31;
  const bf16* p = tile + (rowBase + (lane & 15)) * stride + kOff + ((lane & 16) ? 8 : 0);
  v8bf lo = *(const v8bf*)p;
  v8bf hi = *(const v8bf*)(p + 16);
  v16bf f;
#pragma unroll
  for (int i = 0; i < 8; ++i) { f[i] = lo[i]; f[i + 8] = hi[i]; }
  return f;
}
__device__ inline v8f wmma_bf16(v16bf a, v16bf b, v8f c) {
  return __builtin_amdgcn_wmma_f32_16x16x32_bf16(false, a, false, b, (short)0, c, false, false);
}

#if USE_TDM
typedef unsigned u32x4 __attribute__((ext_vector_type(4)));
typedef int      i32x4 __attribute__((ext_vector_type(4)));
typedef int      i32x8 __attribute__((ext_vector_type(8)));

// Generic LDS pointer -> LDS byte offset (ISA 10.2: LDS aperture uses addr[31:0]).
__device__ inline unsigned lds_addr_of(const void* p) {
  return (unsigned)(unsigned long long)p;
}

// TDM D# for a 2D bf16 tile load: tile_dim0=32 elems (64B rows), tile_dim1=128 rows,
// LDS pad: 4 DWORDs (16B) after every 16 DWORDs (64B) -> LDS row stride 40 elems.
// dim0_rem / rows are remaining tensor extents from the tile origin (HW zero-fills OOB).
__device__ inline void tdm_load_tile(unsigned ldsAddr, const void* g,
                                     unsigned dim0_rem, unsigned rows, unsigned strideK) {
  unsigned long long ga = (unsigned long long)g;
  u32x4 g0;
  g0[0] = 1u;                                        // count=1, no gather
  g0[1] = ldsAddr;                                   // lds_addr (bytes)
  g0[2] = (unsigned)ga;                              // global_addr[31:0]
  g0[3] = (unsigned)((ga >> 32) & 0x1FFFFFFu)        // global_addr[56:32]
          | (2u << 30);                              // type=2 ("image")
  i32x8 g1;
  g1[0] = (int)((1u << 16)                           // data_size = 2 bytes
                | (1u << 20)                         // pad_enable
                | (3u << 22)                         // pad_interval: 16 DWORDs
                | (3u << 25));                       // pad_amount: 4 DWORDs
  g1[1] = (int)((dim0_rem & 0xFFFFu) << 16);         // tensor_dim0[15:0]
  g1[2] = (int)(((dim0_rem >> 16) & 0xFFFFu) | ((rows & 0xFFFFu) << 16));
  g1[3] = (int)(((rows >> 16) & 0xFFFFu) | (32u << 16));   // tile_dim0 = 32
  g1[4] = (int)128;                                  // tile_dim1 = 128, tile_dim2 = 0
  g1[5] = (int)strideK;                              // tensor_dim0_stride[31:0]
  g1[6] = 0;                                         // stride0[47:32]=0, stride1 lo = 0
  g1[7] = 0;
  i32x4 g2 = {0, 0, 0, 0};
  i32x4 g3 = {0, 0, 0, 0};
  i32x8 g4 = {0, 0, 0, 0, 0, 0, 0, 0};
  __builtin_amdgcn_tensor_load_to_lds(g0, g1, g2, g3, g4, 0);
}
#endif

// ---------------------------------------------------------------------------
// generic bf16 WMMA GEMM:  C[M,N] = epi( A[M,K] @ W[N,K]^T )
// EPI: 0=none 1=+bias 2=+bias,softplus 3=+bias,gelu 4=+resid 5=+bias+resid
// ---------------------------------------------------------------------------
template <typename OutT, int EPI>
__global__ __launch_bounds__(256) void gemm_bf16_kernel(
    const bf16* __restrict__ A, const bf16* __restrict__ W,
    const float* __restrict__ bias, const float* __restrict__ resid,
    OutT* __restrict__ C, int M, int N, int K) {
  int t = threadIdx.x;
  int lane = t & 31, wave = t >> 5;
  int wm = wave >> 2, wn = wave & 3;           // 2x4 wave grid
  int mBlk = blockIdx.y * 128, nBlk = blockIdx.x * 128;

  v8f acc[4][2];
#pragma unroll
  for (int mi = 0; mi < 4; ++mi)
#pragma unroll
    for (int ni = 0; ni < 2; ++ni) acc[mi][ni] = v8f_zero();

#if USE_TDM
  __shared__ __align__(16) bf16 As[2][128 * 40];
  __shared__ __align__(16) bf16 Ws[2][128 * 40];
  const bf16* gAbase = A + (size_t)mBlk * K;
  const bf16* gWbase = W + (size_t)nBlk * K;
  unsigned aRows = (unsigned)(M - mBlk);
  unsigned wRows = (unsigned)(N - nBlk);       // may be <128: TDM zero-fills OOB rows
  if (wave == 0) {
    tdm_load_tile(lds_addr_of(As[0]), gAbase, (unsigned)K, aRows, (unsigned)K);
    tdm_load_tile(lds_addr_of(Ws[0]), gWbase, (unsigned)K, wRows, (unsigned)K);
  }
  for (int k0 = 0; k0 < K; k0 += 32) {
    int cur = (k0 >> 5) & 1;
    __syncthreads();                            // buf[cur^1] readers from prev iter done
    if (wave == 0) {
      if (k0 + 32 < K) {                        // prefetch next tile, wait current pair
        tdm_load_tile(lds_addr_of(As[cur ^ 1]), gAbase + (k0 + 32),
                      (unsigned)(K - k0 - 32), aRows, (unsigned)K);
        tdm_load_tile(lds_addr_of(Ws[cur ^ 1]), gWbase + (k0 + 32),
                      (unsigned)(K - k0 - 32), wRows, (unsigned)K);
        __builtin_amdgcn_s_wait_tensorcnt(2);   // in-order: current 2 loads complete
      } else {
        __builtin_amdgcn_s_wait_tensorcnt(0);
      }
    }
    __syncthreads();                            // publish buf[cur]
    v16bf af[4];
#pragma unroll
    for (int mi = 0; mi < 4; ++mi) af[mi] = load_fragA(As[cur], wm * 64 + mi * 16, 40, 0);
#pragma unroll
    for (int ni = 0; ni < 2; ++ni) {
      v16bf bfr = load_fragB(Ws[cur], wn * 32 + ni * 16, 40, 0);
#pragma unroll
      for (int mi = 0; mi < 4; ++mi) acc[mi][ni] = wmma_bf16(af[mi], bfr, acc[mi][ni]);
    }
  }
#else
  __shared__ __align__(16) bf16 As[128 * 40];
  __shared__ __align__(16) bf16 Ws[128 * 40];
  int arow = t >> 1, acol = (t & 1) * 16;      // 128 rows x 32 cols, 16 elems/thread
  const bf16* gA = A + (size_t)(mBlk + arow) * K + acol;
  int wrow = nBlk + arow;
  const bf16* gW = W + (size_t)wrow * K + acol;
  bool wok = (wrow < N);
  v8bf z8;
#pragma unroll
  for (int i = 0; i < 8; ++i) z8[i] = (bf16)0.0f;
  for (int k0 = 0; k0 < K; k0 += 32) {
    v8bf a0 = *(const v8bf*)(gA + k0);
    v8bf a1 = *(const v8bf*)(gA + k0 + 8);
    v8bf w0 = wok ? *(const v8bf*)(gW + k0) : z8;
    v8bf w1 = wok ? *(const v8bf*)(gW + k0 + 8) : z8;
    __syncthreads();
    *(v8bf*)(As + arow * 40 + acol) = a0;
    *(v8bf*)(As + arow * 40 + acol + 8) = a1;
    *(v8bf*)(Ws + arow * 40 + acol) = w0;
    *(v8bf*)(Ws + arow * 40 + acol + 8) = w1;
    __syncthreads();
    v16bf af[4];
#pragma unroll
    for (int mi = 0; mi < 4; ++mi) af[mi] = load_fragA(As, wm * 64 + mi * 16, 40, 0);
#pragma unroll
    for (int ni = 0; ni < 2; ++ni) {
      v16bf bfr = load_fragB(Ws, wn * 32 + ni * 16, 40, 0);
#pragma unroll
      for (int mi = 0; mi < 4; ++mi) acc[mi][ni] = wmma_bf16(af[mi], bfr, acc[mi][ni]);
    }
  }
#endif

  // epilogue — C layout: VGPR i -> row i (lanes 0-15) / i+8 (lanes 16-31); N = lane&15
#pragma unroll
  for (int mi = 0; mi < 4; ++mi)
#pragma unroll
    for (int ni = 0; ni < 2; ++ni) {
      int col = nBlk + wn * 32 + ni * 16 + (lane & 15);
      if (col < N) {
        float bv = 0.0f;
        if (EPI == 1 || EPI == 2 || EPI == 3 || EPI == 5) bv = bias[col];
#pragma unroll
        for (int i = 0; i < 8; ++i) {
          int row = mBlk + wm * 64 + mi * 16 + i + ((lane & 16) ? 8 : 0);
          float v = acc[mi][ni][i] + bv;
          if (EPI == 2) v = softplus_f(v);
          if (EPI == 3) v = gelu_f(v);
          if (EPI == 4 || EPI == 5) v += resid[(size_t)row * N + col];
          C[(size_t)row * N + col] = (OutT)v;
        }
      }
    }
}

// ---------------------------------------------------------------------------
// LayerNorm over D_MODEL, output bf16
// ---------------------------------------------------------------------------
__global__ __launch_bounds__(256) void ln_bf16_kernel(
    const float* __restrict__ x, const float* __restrict__ g,
    const float* __restrict__ b, bf16* __restrict__ out) {
  __shared__ float red[256];
  int row = blockIdx.x, t = threadIdx.x;
  const float* xr = x + (size_t)row * D_MODEL;
  float v[4], s = 0.0f;
#pragma unroll
  for (int i = 0; i < 4; ++i) { v[i] = xr[t + i * 256]; s += v[i]; }
  red[t] = s; __syncthreads();
  for (int o = 128; o > 0; o >>= 1) { if (t < o) red[t] += red[t + o]; __syncthreads(); }
  float mean = red[0] * (1.0f / D_MODEL);
  __syncthreads();
  float q = 0.0f;
#pragma unroll
  for (int i = 0; i < 4; ++i) { float d = v[i] - mean; q += d * d; }
  red[t] = q; __syncthreads();
  for (int o = 128; o > 0; o >>= 1) { if (t < o) red[t] += red[t + o]; __syncthreads(); }
  float rstd = rsqrtf(red[0] * (1.0f / D_MODEL) + 1e-5f);
#pragma unroll
  for (int i = 0; i < 4; ++i) {
    int c = t + i * 256;
    out[(size_t)row * D_MODEL + c] = (bf16)((v[i] - mean) * rstd * g[c] + b[c]);
  }
}

// ---------------------------------------------------------------------------
// causal depthwise conv (kernel 4) + SiLU; reads xi = xz[:, 0:2048]
// ---------------------------------------------------------------------------
__global__ void conv_silu_kernel(const float* __restrict__ xz,
                                 const float* __restrict__ cw,
                                 const float* __restrict__ cb,
                                 float* __restrict__ xc, bf16* __restrict__ xcb) {
  int idx = blockIdx.x * 256 + threadIdx.x;
  if (idx >= NROWS * D_INNER) return;
  int d = idx & (D_INNER - 1);
  int row = idx >> 11;
  int l = row & (LSEQ - 1);
  float acc = cb[d];
#pragma unroll
  for (int kk = 0; kk < 4; ++kk) {
    int lp = l - 3 + kk;
    if (lp >= 0) acc += xz[(size_t)(row - 3 + kk) * (2 * D_INNER) + d] * cw[d * 4 + kk];
  }
  acc = silu_f(acc);
  xc[idx] = acc;
  xcb[idx] = (bf16)acc;
}

// ---------------------------------------------------------------------------
// selective scan: one thread per (b, d); 16 states in registers
// ---------------------------------------------------------------------------
__global__ __launch_bounds__(256) void scan_kernel(
    const float* __restrict__ delta, const float* __restrict__ u,
    const float* __restrict__ xdbl, const float* __restrict__ A_log,
    const float* __restrict__ Dp, float* __restrict__ y) {
  int b = blockIdx.y;
  int d = blockIdx.x * 256 + threadIdx.x;
  float Ar[16], h[16];
#pragma unroll
  for (int n = 0; n < 16; ++n) { Ar[n] = -expf(A_log[d * 16 + n]); h[n] = 0.0f; }
  float Dd = Dp[d];
  __shared__ float Bs[16], Cs[16];
  for (int l = 0; l < LSEQ; ++l) {
    int row = b * LSEQ + l;
    __syncthreads();
    if (threadIdx.x < 32) {
      float val = xdbl[(size_t)row * XDBL_N + DT_RANK + threadIdx.x];
      if (threadIdx.x < 16) Bs[threadIdx.x] = val;
      else Cs[threadIdx.x - 16] = val;
    }
    __syncthreads();
    float dt = delta[(size_t)row * D_INNER + d];
    float uu = u[(size_t)row * D_INNER + d];
    float du = dt * uu, accv = 0.0f;
#pragma unroll
    for (int n = 0; n < 16; ++n) {
      h[n] = expf(dt * Ar[n]) * h[n] + du * Bs[n];
      accv += h[n] * Cs[n];
    }
    y[(size_t)row * D_INNER + d] = accv + uu * Dd;
  }
}

// y *= silu(z), cast bf16; z = xz[:, 2048 + d]
__global__ void gate_kernel(const float* __restrict__ y, const float* __restrict__ xz,
                            bf16* __restrict__ out) {
  int idx = blockIdx.x * 256 + threadIdx.x;
  if (idx >= NROWS * D_INNER) return;
  int d = idx & (D_INNER - 1);
  int row = idx >> 11;
  float z = xz[(size_t)row * (2 * D_INNER) + D_INNER + d];
  out[idx] = (bf16)(y[idx] * silu_f(z));
}

__global__ void f32_to_bf16_kernel(const float* __restrict__ in, bf16* __restrict__ out, int n) {
  int i = blockIdx.x * 256 + threadIdx.x;
  if (i < n) out[i] = (bf16)in[i];
}

__global__ void dt_extract_kernel(const float* __restrict__ xdbl, bf16* __restrict__ out) {
  int i = blockIdx.x * 256 + threadIdx.x;
  if (i >= NROWS * DT_RANK) return;
  int r = i >> 6, c = i & 63;
  out[i] = (bf16)xdbl[(size_t)r * XDBL_N + c];
}

// ---------------------------------------------------------------------------
// fused flash attention: grid (L/64, heads, B), 128 threads (4 waves).
// Each wave owns 16 q rows; online softmax; WMMA for S=QK^T and O+=P*V.
// ---------------------------------------------------------------------------
__global__ __launch_bounds__(128) void attn_kernel(const bf16* __restrict__ q,
                                                   const bf16* __restrict__ k,
                                                   const bf16* __restrict__ v,
                                                   bf16* __restrict__ o) {
  __shared__ __align__(16) bf16 Qs[64 * 72];
  __shared__ __align__(16) bf16 Ks[32 * 72];
  __shared__ __align__(16) bf16 VTs[64 * 40];
  __shared__ __align__(16) bf16 Ps[4 * 16 * 40];
  int t = threadIdx.x;
  int lane = t & 31, wave = t >> 5;
  int qbase = blockIdx.z * LSEQ + blockIdx.x * 64;
  int kvbase = blockIdx.z * LSEQ;
  int hbase = blockIdx.y * 64;

  { // stage Q block [64 x 64]
    int r = t >> 1, c = (t & 1) * 32;
    const bf16* gp = q + (size_t)(qbase + r) * D_MODEL + hbase + c;
    bf16* sp = Qs + r * 72 + c;
#pragma unroll
    for (int i = 0; i < 4; ++i) *(v8bf*)(sp + i * 8) = *(const v8bf*)(gp + i * 8);
  }

  float mrow[8], lrow[8];
  v8f oacc[4];
#pragma unroll
  for (int i = 0; i < 8; ++i) { mrow[i] = -1e30f; lrow[i] = 0.0f; }
#pragma unroll
  for (int n = 0; n < 4; ++n) oacc[n] = v8f_zero();

  for (int kb = 0; kb < LSEQ; kb += 32) {
    __syncthreads();
    { // stage K tile [32 x 64]
      int r = t >> 2, c = (t & 3) * 16;
      const bf16* gp = k + (size_t)(kvbase + kb + r) * D_MODEL + hbase + c;
      bf16* sp = Ks + r * 72 + c;
      *(v8bf*)sp = *(const v8bf*)gp;
      *(v8bf*)(sp + 8) = *(const v8bf*)(gp + 8);
    }
    { // stage V^T tile [64 x 32]
      int j = t & 31, dbase = (t >> 5) * 16;
      const bf16* gp = v + (size_t)(kvbase + kb + j) * D_MODEL + hbase + dbase;
#pragma unroll
      for (int i = 0; i < 16; ++i) VTs[(dbase + i) * 40 + j] = gp[i];
    }
    __syncthreads();

    // S = Q(16x64) @ K^T -> two 16x16 tiles
    v8f s0 = v8f_zero(), s1 = v8f_zero();
#pragma unroll
    for (int ks = 0; ks < 64; ks += 32) {
      v16bf aq = load_fragA(Qs, wave * 16, 72, ks);
      v16bf b0 = load_fragB(Ks, 0, 72, ks);
      v16bf b1 = load_fragB(Ks, 16, 72, ks);
      s0 = wmma_bf16(aq, b0, s0);
      s1 = wmma_bf16(aq, b1, s1);
    }
    // online softmax (row = VGPR index; 16 lanes in a half hold one row's cols)
    float p0[8], p1[8];
#pragma unroll
    for (int i = 0; i < 8; ++i) {
      float a = s0[i] * 0.125f, c = s1[i] * 0.125f;
      float mx = fmaxf(a, c);
      for (int m = 1; m < 16; m <<= 1) mx = fmaxf(mx, __shfl_xor(mx, m, 32));
      float mnew = fmaxf(mrow[i], mx);
      float alpha = expf(mrow[i] - mnew);
      p0[i] = expf(a - mnew);
      p1[i] = expf(c - mnew);
      float rs = p0[i] + p1[i];
      for (int m = 1; m < 16; m <<= 1) rs += __shfl_xor(rs, m, 32);
      lrow[i] = lrow[i] * alpha + rs;
      mrow[i] = mnew;
#pragma unroll
      for (int n = 0; n < 4; ++n) oacc[n][i] *= alpha;
    }
    // transpose P through per-wave LDS into an A fragment
    bf16* pw = Ps + wave * 16 * 40;
    int prow = (lane & 16) ? 8 : 0;
#pragma unroll
    for (int i = 0; i < 8; ++i) {
      pw[(prow + i) * 40 + (lane & 15)] = (bf16)p0[i];
      pw[(prow + i) * 40 + 16 + (lane & 15)] = (bf16)p1[i];
    }
    __syncthreads();
    v16bf pf = load_fragA(pw, 0, 40, 0);
#pragma unroll
    for (int n = 0; n < 4; ++n) {
      v16bf bv = load_fragB(VTs, n * 16, 40, 0);
      oacc[n] = wmma_bf16(pf, bv, oacc[n]);
    }
  }

#pragma unroll
  for (int n = 0; n < 4; ++n)
#pragma unroll
    for (int i = 0; i < 8; ++i) {
      int row = qbase + wave * 16 + i + ((lane & 16) ? 8 : 0);
      int col = hbase + n * 16 + (lane & 15);
      o[(size_t)row * D_MODEL + col] = (bf16)(oacc[n][i] / lrow[i]);
    }
}

// ---------------------------------------------------------------------------
// host orchestration
// ---------------------------------------------------------------------------
extern "C" void kernel_launch(void* const* d_in, const int* in_sizes, int n_in,
                              void* d_out, int out_size, void* d_ws, size_t ws_size,
                              hipStream_t stream) {
  (void)in_sizes; (void)n_in; (void)out_size; (void)ws_size;
  const float* x        = (const float*)d_in[0];
  const float* memory   = (const float*)d_in[1];
  const float* ln1_g    = (const float*)d_in[2];
  const float* ln1_b    = (const float*)d_in[3];
  const float* in_projw = (const float*)d_in[4];
  const float* conv_w   = (const float*)d_in[5];
  const float* conv_b   = (const float*)d_in[6];
  const float* xproj_w  = (const float*)d_in[7];
  const float* dt_w     = (const float*)d_in[8];
  const float* dt_b     = (const float*)d_in[9];
  const float* A_log    = (const float*)d_in[10];
  const float* Dp       = (const float*)d_in[11];
  const float* outp_w   = (const float*)d_in[12];
  const float* ln2_g    = (const float*)d_in[13];
  const float* ln2_b    = (const float*)d_in[14];
  const float* wq       = (const float*)d_in[15];
  const float* bq       = (const float*)d_in[16];
  const float* wk       = (const float*)d_in[17];
  const float* bk       = (const float*)d_in[18];
  const float* wv       = (const float*)d_in[19];
  const float* bv       = (const float*)d_in[20];
  const float* wo       = (const float*)d_in[21];
  const float* bo       = (const float*)d_in[22];
  const float* ln3_g    = (const float*)d_in[23];
  const float* ln3_b    = (const float*)d_in[24];
  const float* ff_w1    = (const float*)d_in[25];
  const float* ff_b1    = (const float*)d_in[26];
  const float* ff_w2    = (const float*)d_in[27];
  const float* ff_b2    = (const float*)d_in[28];
  float* out = (float*)d_out;

  char* base = (char*)d_ws;
  size_t off = 0;
  auto alloc = [&](size_t bytes) -> void* {
    void* p = base + off;
    off += (bytes + 255) & ~(size_t)255;
    return p;
  };
  // bf16 weights
  bf16* wip  = (bf16*)alloc((size_t)2 * D_INNER * D_MODEL * 2);
  bf16* wxp  = (bf16*)alloc((size_t)XDBL_N * D_INNER * 2);
  bf16* wdt  = (bf16*)alloc((size_t)D_INNER * DT_RANK * 2);
  bf16* wop  = (bf16*)alloc((size_t)D_MODEL * D_INNER * 2);
  bf16* wqb  = (bf16*)alloc((size_t)D_MODEL * D_MODEL * 2);
  bf16* wkb  = (bf16*)alloc((size_t)D_MODEL * D_MODEL * 2);
  bf16* wvb  = (bf16*)alloc((size_t)D_MODEL * D_MODEL * 2);
  bf16* wob  = (bf16*)alloc((size_t)D_MODEL * D_MODEL * 2);
  bf16* wf1  = (bf16*)alloc((size_t)4 * D_MODEL * D_MODEL * 2);
  bf16* wf2  = (bf16*)alloc((size_t)4 * D_MODEL * D_MODEL * 2);
  bf16* membf = (bf16*)alloc((size_t)NROWS * D_MODEL * 2);
  // activations
  bf16*  ln1bf = (bf16*)alloc((size_t)NROWS * D_MODEL * 2);
  float* xz    = (float*)alloc((size_t)NROWS * 2 * D_INNER * 4);
  float* xcf   = (float*)alloc((size_t)NROWS * D_INNER * 4);
  bf16*  xcbf  = (bf16*)alloc((size_t)NROWS * D_INNER * 2);
  float* xdbl  = (float*)alloc((size_t)NROWS * XDBL_N * 4);
  bf16*  dtbf  = (bf16*)alloc((size_t)NROWS * DT_RANK * 2);
  float* delta = (float*)alloc((size_t)NROWS * D_INNER * 4);
  float* yf    = (float*)alloc((size_t)NROWS * D_INNER * 4);
  bf16*  ybf   = (bf16*)alloc((size_t)NROWS * D_INNER * 2);
  float* h     = (float*)alloc((size_t)NROWS * D_MODEL * 4);
  bf16*  ln2bf = (bf16*)alloc((size_t)NROWS * D_MODEL * 2);
  bf16*  qbf   = (bf16*)alloc((size_t)NROWS * D_MODEL * 2);
  bf16*  kbf   = (bf16*)alloc((size_t)NROWS * D_MODEL * 2);
  bf16*  vbf   = (bf16*)alloc((size_t)NROWS * D_MODEL * 2);
  bf16*  atbf  = (bf16*)alloc((size_t)NROWS * D_MODEL * 2);
  float* h2    = (float*)alloc((size_t)NROWS * D_MODEL * 4);
  bf16*  ln3bf = (bf16*)alloc((size_t)NROWS * D_MODEL * 2);
  bf16*  ffmid = (bf16*)alloc((size_t)NROWS * 4 * D_MODEL * 2);

  auto cvt = [&](const float* src, bf16* dst, int n) {
    f32_to_bf16_kernel<<<(n + 255) / 256, 256, 0, stream>>>(src, dst, n);
  };
  cvt(in_projw, wip, 2 * D_INNER * D_MODEL);
  cvt(xproj_w,  wxp, XDBL_N * D_INNER);
  cvt(dt_w,     wdt, D_INNER * DT_RANK);
  cvt(outp_w,   wop, D_MODEL * D_INNER);
  cvt(wq, wqb, D_MODEL * D_MODEL);
  cvt(wk, wkb, D_MODEL * D_MODEL);
  cvt(wv, wvb, D_MODEL * D_MODEL);
  cvt(wo, wob, D_MODEL * D_MODEL);
  cvt(ff_w1, wf1, 4 * D_MODEL * D_MODEL);
  cvt(ff_w2, wf2, 4 * D_MODEL * D_MODEL);
  cvt(memory, membf, NROWS * D_MODEL);

  dim3 blk(256);
  // ---- Mamba branch ----
  ln_bf16_kernel<<<NROWS, blk, 0, stream>>>(x, ln1_g, ln1_b, ln1bf);
  gemm_bf16_kernel<float, 0><<<dim3(2 * D_INNER / 128, NROWS / 128), blk, 0, stream>>>(
      ln1bf, wip, nullptr, nullptr, xz, NROWS, 2 * D_INNER, D_MODEL);
  conv_silu_kernel<<<(NROWS * D_INNER + 255) / 256, blk, 0, stream>>>(xz, conv_w, conv_b, xcf, xcbf);
  gemm_bf16_kernel<float, 0><<<dim3(1, NROWS / 128), blk, 0, stream>>>(
      xcbf, wxp, nullptr, nullptr, xdbl, NROWS, XDBL_N, D_INNER);
  dt_extract_kernel<<<(NROWS * DT_RANK + 255) / 256, blk, 0, stream>>>(xdbl, dtbf);
  gemm_bf16_kernel<float, 2><<<dim3(D_INNER / 128, NROWS / 128), blk, 0, stream>>>(
      dtbf, wdt, dt_b, nullptr, delta, NROWS, D_INNER, DT_RANK);
  scan_kernel<<<dim3(D_INNER / 256, NB), blk, 0, stream>>>(delta, xcf, xdbl, A_log, Dp, yf);
  gate_kernel<<<(NROWS * D_INNER + 255) / 256, blk, 0, stream>>>(yf, xz, ybf);
  gemm_bf16_kernel<float, 4><<<dim3(D_MODEL / 128, NROWS / 128), blk, 0, stream>>>(
      ybf, wop, nullptr, x, h, NROWS, D_MODEL, D_INNER);

  // ---- cross attention ----
  ln_bf16_kernel<<<NROWS, blk, 0, stream>>>(h, ln2_g, ln2_b, ln2bf);
  gemm_bf16_kernel<bf16, 1><<<dim3(D_MODEL / 128, NROWS / 128), blk, 0, stream>>>(
      ln2bf, wqb, bq, nullptr, qbf, NROWS, D_MODEL, D_MODEL);
  gemm_bf16_kernel<bf16, 1><<<dim3(D_MODEL / 128, NROWS / 128), blk, 0, stream>>>(
      membf, wkb, bk, nullptr, kbf, NROWS, D_MODEL, D_MODEL);
  gemm_bf16_kernel<bf16, 1><<<dim3(D_MODEL / 128, NROWS / 128), blk, 0, stream>>>(
      membf, wvb, bv, nullptr, vbf, NROWS, D_MODEL, D_MODEL);
  attn_kernel<<<dim3(LSEQ / 64, 16, NB), dim3(128), 0, stream>>>(qbf, kbf, vbf, atbf);
  gemm_bf16_kernel<float, 5><<<dim3(D_MODEL / 128, NROWS / 128), blk, 0, stream>>>(
      atbf, wob, bo, h, h2, NROWS, D_MODEL, D_MODEL);

  // ---- FFN ----
  ln_bf16_kernel<<<NROWS, blk, 0, stream>>>(h2, ln3_g, ln3_b, ln3bf);
  gemm_bf16_kernel<bf16, 3><<<dim3(4 * D_MODEL / 128, NROWS / 128), blk, 0, stream>>>(
      ln3bf, wf1, ff_b1, nullptr, ffmid, NROWS, 4 * D_MODEL, D_MODEL);
  gemm_bf16_kernel<float, 5><<<dim3(D_MODEL / 128, NROWS / 128), blk, 0, stream>>>(
      ffmid, wf2, ff_b2, h2, out, NROWS, D_MODEL, 4 * D_MODEL);
}
